// MoEConnectionProcessor_67164698574981
// MI455X (gfx1250) — matrix operand order repaired and checked by reading, not compiled, not run on hardware
//
#include <hip/hip_runtime.h>
#include <math.h>

// ---------------------------------------------------------------------------
// CDNA5 (gfx1250) fused MoE connection processor.
// - All matrix math on v_wmma_f32_16x16x32_bf16 (f32 accumulate).
// - neighbor_states streamed via GLOBAL_LOAD_ASYNC_TO_LDS_B128 (ASYNCcnt),
//   double-buffered so DMA of row b+1 overlaps WMMA of row b.
// - cs @ W_msg_top hoisted out of the per-neighbor GEMM: computed once per
//   16-row batch tile (8 WMMAs) instead of per neighbor row.
// - Split-counter waits: s_wait_asynccnt / s_wait_dscnt.
// ---------------------------------------------------------------------------

typedef __attribute__((ext_vector_type(16))) __bf16 v16bf;
typedef __attribute__((ext_vector_type(8)))  float  v8f;
typedef __attribute__((ext_vector_type(16))) unsigned short v16u;
typedef __attribute__((ext_vector_type(8)))  unsigned short v8u;
typedef __attribute__((ext_vector_type(4)))  int v4i;

#define AS1 __attribute__((address_space(1)))
#define AS3 __attribute__((address_space(3)))

#define NW   8            // waves per workgroup
#define NTHR (NW * 32)
#define XROW 72           // bf16 16-row tile stride (elems) = 144B, 16B-aligned
#define SROW 68           // f32 staging row stride (elems) = 272B, 16B-aligned

// ---- shared weight region (bytes) ----
#define OFF_WLB   0u      // W_local   B-layout bf16 (16384)
#define OFF_WUB   16384u  // W_upd     (16384)
#define OFF_WDB   32768u  // W_dcnf    (16384)
#define OFF_WMTB  49152u  // W_msg[0:64]   top half (8192)
#define OFF_WMBB  57344u  // W_msg[64:128] bottom half (8192)
#define OFF_WFB   65536u  // W_fcnf    (8192)
#define OFF_WG1B  73728u  // W_g1      (8192)
#define OFF_WG2   81920u  // W_g2 f32  (512)
#define OFF_BIAS  82432u  // biases f32 (1536)
#define OFF_PW    83968u

// ---- per-wave region (bytes) ----
#define PO_STG0   0u      // ns staging buf 0, f32 [26][SROW] (7072)
#define PO_STG1   7072u   // ns staging buf 1 (7072)
#define PO_CSM    14144u  // cs @ W_msg_top, f32 [16][68] (4352)
#define PO_CST    18496u  // cs tile bf16 [16][XROW] (2304)
#define PO_ALT    20800u  // agg_local tile
#define PO_AFT    23104u  // agg_func tile
#define PO_ADT    25408u  // agg_dist tile
#define PO_MNT    27712u  // mean tile
#define PW_BYTES  30016u
// phase-B scratch aliases the dead staging buffers:
#define PO_HT     0u      // h tile bf16 (2304)
#define PO_GSC    2304u   // g1 relayout f32 [16][32] (2048)
#define PO_GW     4352u   // gate weights f32 [16][4] (256)

#define SMEM_BYTES (OFF_PW + NW * PW_BYTES)   // 324096 <= 327680 (320KB/WGP)

__device__ __forceinline__ unsigned short f2bf(float f) {
  unsigned u = __builtin_bit_cast(unsigned, f);
  u += 0x7fffu + ((u >> 16) & 1u);            // round-to-nearest-even
  return (unsigned short)(u >> 16);
}
__device__ __forceinline__ float bf2f(unsigned short u) {
  return __builtin_bit_cast(float, ((unsigned)u) << 16);
}
__device__ __forceinline__ v8f v8zero() {
  v8f z = {0.f, 0.f, 0.f, 0.f, 0.f, 0.f, 0.f, 0.f};
  return z;
}
__device__ __forceinline__ void wait_ds() {
  asm volatile("s_wait_dscnt 0" ::: "memory");
}
__device__ __forceinline__ void wait_async0() {
  asm volatile("s_wait_asynccnt 0" ::: "memory");
}
__device__ __forceinline__ void wait_async13() {
  asm volatile("s_wait_asynccnt 13" ::: "memory");
}

// Async DMA one batch row's neighbors (K x 64 f32, row-major) into LDS
// staging ([K][SROW] f32). 16B chunks, 32 lanes per instruction, ASYNCcnt.
__device__ __forceinline__ void issue_ns_async(const float* __restrict__ src,
                                               char* dstBase, int lane, int K) {
  const int nch = K * 16;                    // 16-byte chunks
  const int iters = (nch + 31) >> 5;
  for (int it = 0; it < iters; ++it) {
    const int i = it * 32 + lane;
    if (i < nch) {
      const int r = i >> 4, c = i & 15;
      void* gp = (void*)(src + (size_t)r * 64 + (c << 2));
      void* lp = (void*)(dstBase + r * (SROW * 4) + (c << 4));
      __builtin_amdgcn_global_load_async_to_lds_b128(
          (AS1 v4i*)gp, (AS3 v4i*)lp, 0, 0);
    }
  }
}

// Stage a row-major f32 [Kin x Nout] weight into WMMA B-operand bf16 layout.
__device__ __forceinline__ void stageB(const float* __restrict__ src,
                                       unsigned short* dst,
                                       int Kin, int Nout, int tid, int nthr) {
  const int NT = Nout >> 4;
  const int total = Kin * Nout;
  for (int i = tid; i < total; i += nthr) {
    const int j   = i & 15;
    const int l   = (i >> 4) & 31;
    const int qnt = i >> 9;
    const int q   = qnt / NT;
    const int nt  = qnt - q * NT;
    const int kk  = q * 32 + ((j >> 3) << 4) + ((l >> 4) << 3) + (j & 7);
    const int nn  = nt * 16 + (l & 15);
    dst[i] = f2bf(src[kk * Nout + nn]);
  }
}

// M=16 GEMM from bf16 tiles (tLo = K cols 0..63, tHi = 64..127).
template <int KQ, int NT>
__device__ __forceinline__ void gemm16(const unsigned short* tLo,
                                       const unsigned short* tHi,
                                       const unsigned short* Wb,
                                       int lane, v8f* acc) {
  const int row = lane & 15;
  const int kh  = (lane >> 4) << 3;
  #pragma unroll
  for (int q = 0; q < KQ; ++q) {
    const unsigned short* t  = (q >= 2) ? tHi : tLo;
    const unsigned short* ap = t + row * XROW + (q & 1) * 32 + kh;
    v8u lo = *(const v8u*)ap;
    v8u hi = *(const v8u*)(ap + 16);
    v16u a;
    #pragma unroll
    for (int i = 0; i < 8; ++i) { a[i] = lo[i]; a[i + 8] = hi[i]; }
    v16bf af = __builtin_bit_cast(v16bf, a);
    #pragma unroll
    for (int nt = 0; nt < NT; ++nt) {
      v16u bu16 = *(const v16u*)(Wb + (((q * NT + nt) * 32) + lane) * 16);
      v16bf bv  = __builtin_bit_cast(v16bf, bu16);
      acc[nt] = __builtin_amdgcn_wmma_f32_16x16x32_bf16(
          false, af, false, bv, (short)0, acc[nt], false, false);
    }
  }
}

__global__ void __launch_bounds__(NTHR, 1)
moe_wmma_kernel(const float* __restrict__ cs_g,
                const float* __restrict__ ns_g,
                const int*   __restrict__ tier_g,
                const float* __restrict__ Wl_g,  const float* __restrict__ bl_g,
                const float* __restrict__ Wm_g,  const float* __restrict__ bm_g,
                const float* __restrict__ Wu_g,  const float* __restrict__ bu_g,
                const float* __restrict__ Wf_g,  const float* __restrict__ bf_g,
                const float* __restrict__ Wd_g,  const float* __restrict__ bd_g,
                const float* __restrict__ Wg1_g, const float* __restrict__ bg1_g,
                const float* __restrict__ Wg2_g, const float* __restrict__ bg2_g,
                float* __restrict__ out_g, int B, int K) {
  extern __shared__ char smem[];
  unsigned short* WLB  = (unsigned short*)(smem + OFF_WLB);
  unsigned short* WUB  = (unsigned short*)(smem + OFF_WUB);
  unsigned short* WDB  = (unsigned short*)(smem + OFF_WDB);
  unsigned short* WMTB = (unsigned short*)(smem + OFF_WMTB);
  unsigned short* WMBB = (unsigned short*)(smem + OFF_WMBB);
  unsigned short* WFB  = (unsigned short*)(smem + OFF_WFB);
  unsigned short* WG1B = (unsigned short*)(smem + OFF_WG1B);
  float* wg2  = (float*)(smem + OFF_WG2);
  float* bias = (float*)(smem + OFF_BIAS);

  const int tid  = threadIdx.x;
  const int lane = tid & 31;
  const int wave = tid >> 5;

  // ---- stage weights once per workgroup ----
  stageB(Wl_g,       WLB,  128, 64, tid, NTHR);
  stageB(Wu_g,       WUB,  128, 64, tid, NTHR);
  stageB(Wd_g,       WDB,  128, 64, tid, NTHR);
  stageB(Wm_g,       WMTB,  64, 64, tid, NTHR);    // W_msg rows 0..63 (cs)
  stageB(Wm_g + 64 * 64, WMBB, 64, 64, tid, NTHR); // W_msg rows 64..127 (ns)
  stageB(Wf_g,       WFB,   64, 64, tid, NTHR);
  stageB(Wg1_g,      WG1B, 128, 32, tid, NTHR);
  for (int i = tid; i < 96; i += NTHR) wg2[i] = Wg2_g[i];
  if (tid < 64) {
    bias[tid]       = bl_g[tid];
    bias[64 + tid]  = bm_g[tid];
    bias[128 + tid] = bu_g[tid];
    bias[192 + tid] = bf_g[tid];
    bias[256 + tid] = bd_g[tid];
  }
  if (tid < 32) bias[320 + tid] = bg1_g[tid];
  if (tid < 3)  bias[352 + tid] = bg2_g[tid];
  __syncthreads();

  const float* biasL  = bias;
  const float* biasM  = bias + 64;
  const float* biasU  = bias + 128;
  const float* biasF  = bias + 192;
  const float* biasD  = bias + 256;
  const float* biasG1 = bias + 320;
  const float* biasG2 = bias + 352;

  char* pw = smem + OFF_PW + (unsigned)wave * PW_BYTES;
  char* stg0 = pw + PO_STG0;
  char* stg1 = pw + PO_STG1;
  float* csM = (float*)(pw + PO_CSM);
  unsigned short* csT = (unsigned short*)(pw + PO_CST);
  unsigned short* aLT = (unsigned short*)(pw + PO_ALT);
  unsigned short* aFT = (unsigned short*)(pw + PO_AFT);
  unsigned short* aDT = (unsigned short*)(pw + PO_ADT);
  unsigned short* mnT = (unsigned short*)(pw + PO_MNT);
  unsigned short* hT  = (unsigned short*)(pw + PO_HT);  // aliases stg0 (phase B)
  float* gsc = (float*)(pw + PO_GSC);                   // aliases stg0
  float* gw4 = (float*)(pw + PO_GW);                    // aliases stg0

  const int bbase = (blockIdx.x * NW + wave) * 16;
  const int row16 = lane & 15;
  const int kh    = (lane >> 4) << 3;
  const bool pipe13 = (((K * 16 + 31) >> 5) == 13);     // K==26 fast path

  // ===== Phase 0: stage cs tile, compute cs @ W_msg_top once per 16 rows ====
  for (int bi = 0; bi < 16; ++bi) {
    const int b = bbase + bi;
    if (b >= B) break;
    float2 csv = *(const float2*)(cs_g + (size_t)b * 64 + 2 * lane);
    *(unsigned*)(csT + bi * XROW + 2 * lane) =
        (unsigned)f2bf(csv.x) | ((unsigned)f2bf(csv.y) << 16);
  }
  wait_ds();
  {
    v8f a4[4];
    #pragma unroll
    for (int nt = 0; nt < 4; ++nt) a4[nt] = v8zero();
    gemm16<2, 4>(csT, csT, WMTB, lane, a4);
    #pragma unroll
    for (int nt = 0; nt < 4; ++nt)
      #pragma unroll
      for (int j = 0; j < 8; ++j)
        csM[(kh + j) * SROW + nt * 16 + row16] = a4[nt][j];
  }
  wait_ds();

  // ===== Phase A: async-pipelined per-row message GEMM (ns-only A) =====
  if (bbase < B)
    issue_ns_async(ns_g + (size_t)bbase * K * 64, stg0, lane, K);

  for (int bi = 0; bi < 16; ++bi) {
    const int b = bbase + bi;
    if (b >= B) break;
    char* cur = (bi & 1) ? stg1 : stg0;
    char* nxt = (bi & 1) ? stg0 : stg1;
    const bool more = (bi + 1 < 16) && (b + 1 < B);
    if (more)
      issue_ns_async(ns_g + (size_t)(b + 1) * K * 64, nxt, lane, K);
    if (more && pipe13) wait_async13(); else wait_async0();

    // tier masks
    int t = (lane < K) ? tier_g[(size_t)b * K + lane] : 3;
    unsigned m0 = (t == 0) ? (1u << lane) : 0u;
    unsigned m1 = (t == 1) ? (1u << lane) : 0u;
    unsigned m2 = (t == 2) ? (1u << lane) : 0u;
    #pragma unroll
    for (int off = 16; off > 0; off >>= 1) {
      m0 |= __shfl_xor(m0, off, 32);
      m1 |= __shfl_xor(m1, off, 32);
      m2 |= __shfl_xor(m2, off, 32);
    }

    // masked column sums from f32 staging
    const float* curF = (const float*)cur;
    float sa0 = 0.f, sa1 = 0.f, sl0 = 0.f, sl1 = 0.f, sd0 = 0.f, sd1 = 0.f;
    for (int r = 0; r < K; ++r) {
      float2 v = *(const float2*)(curF + r * SROW + 2 * lane);
      sa0 += v.x; sa1 += v.y;
      if ((m0 >> r) & 1u) { sl0 += v.x; sl1 += v.y; }
      if ((m2 >> r) & 1u) { sd0 += v.x; sd1 += v.y; }
    }

    // ns @ W_msg_bot: A = [32 x 64] built from f32 staging with inline cvt
    v8f acc[2][4];
    #pragma unroll
    for (int m = 0; m < 2; ++m)
      #pragma unroll
      for (int nt = 0; nt < 4; ++nt) acc[m][nt] = v8zero();
    #pragma unroll
    for (int q = 0; q < 2; ++q) {
      v16bf afr[2];
      #pragma unroll
      for (int m = 0; m < 2; ++m) {
        const float* ap = curF + (m * 16 + row16) * SROW + q * 32 + kh;
        float4 c0 = *(const float4*)(ap);
        float4 c1 = *(const float4*)(ap + 4);
        float4 c2 = *(const float4*)(ap + 16);
        float4 c3 = *(const float4*)(ap + 20);
        v16u a;
        a[0]  = f2bf(c0.x); a[1]  = f2bf(c0.y); a[2]  = f2bf(c0.z); a[3]  = f2bf(c0.w);
        a[4]  = f2bf(c1.x); a[5]  = f2bf(c1.y); a[6]  = f2bf(c1.z); a[7]  = f2bf(c1.w);
        a[8]  = f2bf(c2.x); a[9]  = f2bf(c2.y); a[10] = f2bf(c2.z); a[11] = f2bf(c2.w);
        a[12] = f2bf(c3.x); a[13] = f2bf(c3.y); a[14] = f2bf(c3.z); a[15] = f2bf(c3.w);
        afr[m] = __builtin_bit_cast(v16bf, a);
      }
      #pragma unroll
      for (int nt = 0; nt < 4; ++nt) {
        v16u bu16 = *(const v16u*)(WMBB + ((q * 4 + nt) * 32 + lane) * 16);
        v16bf bv = __builtin_bit_cast(v16bf, bu16);
        #pragma unroll
        for (int m = 0; m < 2; ++m)
          acc[m][nt] = __builtin_amdgcn_wmma_f32_16x16x32_bf16(
              false, afr[m], false, bv, (short)0, acc[m][nt], false, false);
      }
    }

    // masked mean of relu(cs-term + ns-term + b_msg) over neighbor rows
    const float invCF = 1.f / fmaxf((float)__popc(m1), 1.f);
    #pragma unroll
    for (int nt = 0; nt < 4; ++nt) {
      const int n = nt * 16 + row16;
      const float base = biasM[n] + csM[bi * SROW + n];
      float s = 0.f;
      #pragma unroll
      for (int m = 0; m < 2; ++m)
        #pragma unroll
        for (int j = 0; j < 8; ++j) {
          const int r = m * 16 + kh + j;
          float v = fmaxf(acc[m][nt][j] + base, 0.f);
          if ((m1 >> r) & 1u) s += v;
        }
      s += __shfl_xor(s, 16, 32);
      if (lane < 16) aFT[bi * XROW + n] = f2bf(s * invCF);
    }

    const float invCL = 1.f / fmaxf((float)__popc(m0), 1.f);
    const float invCD = 1.f / fmaxf((float)__popc(m2), 1.f);
    const float invK  = 1.f / (float)K;
    *(unsigned*)(aLT + bi * XROW + 2 * lane) =
        (unsigned)f2bf(sl0 * invCL) | ((unsigned)f2bf(sl1 * invCL) << 16);
    *(unsigned*)(aDT + bi * XROW + 2 * lane) =
        (unsigned)f2bf(sd0 * invCD) | ((unsigned)f2bf(sd1 * invCD) << 16);
    *(unsigned*)(mnT + bi * XROW + 2 * lane) =
        (unsigned)f2bf(sa0 * invK) | ((unsigned)f2bf(sa1 * invK) << 16);
  }
  wait_ds();

  // ===== Phase B: batched expert heads, M = 16 batch rows per wave =====
  v8f hl[4], hf[4], hd[4];

  { // local expert: tanh([cs|aggL] @ W_local + b)
    v8f a4[4];
    #pragma unroll
    for (int nt = 0; nt < 4; ++nt) a4[nt] = v8zero();
    gemm16<4, 4>(csT, aLT, WLB, lane, a4);
    #pragma unroll
    for (int nt = 0; nt < 4; ++nt)
      #pragma unroll
      for (int j = 0; j < 8; ++j)
        hl[nt][j] = tanhf(a4[nt][j] + biasL[nt * 16 + row16]);
  }
  { // functional update: tanh([cs|aggF] @ W_upd + b)
    v8f a4[4];
    #pragma unroll
    for (int nt = 0; nt < 4; ++nt) a4[nt] = v8zero();
    gemm16<4, 4>(csT, aFT, WUB, lane, a4);
    #pragma unroll
    for (int nt = 0; nt < 4; ++nt)
      #pragma unroll
      for (int j = 0; j < 8; ++j)
        hf[nt][j] = tanhf(a4[nt][j] + biasU[nt * 16 + row16]);
  }
  // functional CNF: h += dt * tanh(h @ W_fcnf + b), 3 Euler steps
  #pragma unroll
  for (int st = 0; st < 3; ++st) {
    #pragma unroll
    for (int nt = 0; nt < 4; ++nt)
      #pragma unroll
      for (int j = 0; j < 8; ++j)
        hT[(kh + j) * XROW + nt * 16 + row16] = f2bf(hf[nt][j]);
    wait_ds();
    v8f a4[4];
    #pragma unroll
    for (int nt = 0; nt < 4; ++nt) a4[nt] = v8zero();
    gemm16<2, 4>(hT, hT, WFB, lane, a4);
    #pragma unroll
    for (int nt = 0; nt < 4; ++nt)
      #pragma unroll
      for (int j = 0; j < 8; ++j)
        hf[nt][j] += 0.1f * tanhf(a4[nt][j] + biasF[nt * 16 + row16]);
  }
  // distant expert CNF: h0 = cs; h += dt * tanh([h|aggD] @ W_dcnf + b)
  #pragma unroll
  for (int nt = 0; nt < 4; ++nt)
    #pragma unroll
    for (int j = 0; j < 8; ++j)
      hd[nt][j] = bf2f(csT[(kh + j) * XROW + nt * 16 + row16]);
  #pragma unroll
  for (int st = 0; st < 3; ++st) {
    #pragma unroll
    for (int nt = 0; nt < 4; ++nt)
      #pragma unroll
      for (int j = 0; j < 8; ++j)
        hT[(kh + j) * XROW + nt * 16 + row16] = f2bf(hd[nt][j]);
    wait_ds();
    v8f a4[4];
    #pragma unroll
    for (int nt = 0; nt < 4; ++nt) a4[nt] = v8zero();
    gemm16<4, 4>(hT, aDT, WDB, lane, a4);
    #pragma unroll
    for (int nt = 0; nt < 4; ++nt)
      #pragma unroll
      for (int j = 0; j < 8; ++j)
        hd[nt][j] += 0.1f * tanhf(a4[nt][j] + biasD[nt * 16 + row16]);
  }
  { // gating hidden: relu([cs|mean] @ W_g1 + b), N = 32
    v8f a2[2];
    #pragma unroll
    for (int nt = 0; nt < 2; ++nt) a2[nt] = v8zero();
    gemm16<4, 2>(csT, mnT, WG1B, lane, a2);
    #pragma unroll
    for (int nt = 0; nt < 2; ++nt)
      #pragma unroll
      for (int j = 0; j < 8; ++j)
        gsc[(kh + j) * 32 + nt * 16 + row16] =
            fmaxf(a2[nt][j] + biasG1[nt * 16 + row16], 0.f);
  }
  wait_ds();
  if (lane < 16) { // logits + softmax, one batch row per lane
    float z0 = biasG2[0], z1 = biasG2[1], z2 = biasG2[2];
    #pragma unroll
    for (int k2 = 0; k2 < 32; ++k2) {
      float gv = gsc[lane * 32 + k2];
      z0 += gv * wg2[k2 * 3 + 0];
      z1 += gv * wg2[k2 * 3 + 1];
      z2 += gv * wg2[k2 * 3 + 2];
    }
    float zm = fmaxf(z0, fmaxf(z1, z2));
    float e0 = __expf(z0 - zm), e1 = __expf(z1 - zm), e2 = __expf(z2 - zm);
    float inv = 1.f / (e0 + e1 + e2);
    gw4[lane * 4 + 0] = e0 * inv;
    gw4[lane * 4 + 1] = e1 * inv;
    gw4[lane * 4 + 2] = e2 * inv;
  }
  wait_ds();

  float w0j[8], w1j[8], w2j[8];
  #pragma unroll
  for (int j = 0; j < 8; ++j) {
    const int r = kh + j;
    w0j[j] = gw4[r * 4 + 0];
    w1j[j] = gw4[r * 4 + 1];
    w2j[j] = gw4[r * 4 + 2];
  }
  #pragma unroll
  for (int nt = 0; nt < 4; ++nt)
    #pragma unroll
    for (int j = 0; j < 8; ++j) {
      const int b = bbase + kh + j;
      if (b < B)
        out_g[(size_t)b * 64 + nt * 16 + row16] =
            w0j[j] * hl[nt][j] + w1j[j] * hf[nt][j] + w2j[j] * hd[nt][j];
    }
}

extern "C" void kernel_launch(void* const* d_in, const int* in_sizes, int n_in,
                              void* d_out, int out_size, void* d_ws, size_t ws_size,
                              hipStream_t stream) {
  (void)n_in; (void)out_size; (void)d_ws; (void)ws_size;
  const float* cs  = (const float*)d_in[0];
  const float* ns  = (const float*)d_in[1];
  const int*   tr  = (const int*)d_in[2];
  const float* Wl  = (const float*)d_in[3];
  const float* bl  = (const float*)d_in[4];
  const float* Wm  = (const float*)d_in[5];
  const float* bm  = (const float*)d_in[6];
  const float* Wu  = (const float*)d_in[7];
  const float* bu  = (const float*)d_in[8];
  const float* Wf  = (const float*)d_in[9];
  const float* bf  = (const float*)d_in[10];
  const float* Wd  = (const float*)d_in[11];
  const float* bd  = (const float*)d_in[12];
  const float* Wg1 = (const float*)d_in[13];
  const float* bg1 = (const float*)d_in[14];
  const float* Wg2 = (const float*)d_in[15];
  const float* bg2 = (const float*)d_in[16];
  float* out = (float*)d_out;

  const int B = in_sizes[0] / 64;
  const int K = (B > 0) ? in_sizes[2] / B : 0;
  const int blocks = (B + NW * 16 - 1) / (NW * 16);

  moe_wmma_kernel<<<dim3(blocks), dim3(NTHR), SMEM_BYTES, stream>>>(
      cs, ns, tr, Wl, bl, Wm, bm, Wu, bu, Wf, bf, Wd, bd, Wg1, bg1, Wg2, bg2,
      out, B, K);
}
